// CharRNN_68229850464634
// MI455X (gfx1250) — matrix operand order repaired and checked by reading, not compiled, not run on hardware
//
#include <hip/hip_runtime.h>
#include <cstdint>
#include <cstddef>

// ---------- CDNA5 WMMA types ----------
typedef __attribute__((ext_vector_type(16))) __bf16 v16bf;
typedef __attribute__((ext_vector_type(8)))  float  v8f;

__device__ __forceinline__ uint16_t f2bf(float f) {
    uint32_t u = __float_as_uint(f);
    u += 0x7FFFu + ((u >> 16) & 1u);          // round-to-nearest-even
    return (uint16_t)(u >> 16);
}
__device__ __forceinline__ float bf2f(uint16_t h) {
    return __uint_as_float(((uint32_t)h) << 16);
}
__device__ __forceinline__ v16bf pack_frag(uint4 lo, uint4 hi) {
    union { uint4 q[2]; v16bf v; } u;
    u.q[0] = lo; u.q[1] = hi;
    return u.v;
}

// ---------- problem constants ----------
#define VOCAB 10000
#define BB    64
#define SS    256
#define HH    1024
#define EE    256

#define WMMA_BF16(A, B, C) \
    __builtin_amdgcn_wmma_f32_16x16x32_bf16(false, (A), false, (B), (short)0, (C), false, false)

// ======================================================================
// Prep kernels
// ======================================================================
// LDS-tiled fp32 -> bf16 transpose:  dst[n][k] = bf16(src[k][n])
__global__ void k_transpose_bf16(const float* __restrict__ src,
                                 uint16_t* __restrict__ dst, int K, int N) {
    __shared__ float tile[32][33];
    int k0 = blockIdx.x * 32, n0 = blockIdx.y * 32;
#pragma unroll
    for (int i = 0; i < 4; ++i) {
        int k = k0 + threadIdx.y + i * 8;
        int n = n0 + threadIdx.x;
        if (k < K && n < N) tile[threadIdx.y + i * 8][threadIdx.x] = src[(size_t)k * N + n];
    }
    __syncthreads();
#pragma unroll
    for (int i = 0; i < 4; ++i) {
        int n = n0 + threadIdx.y + i * 8;
        int k = k0 + threadIdx.x;
        if (n < N && k < K) dst[(size_t)n * K + k] = f2bf(tile[threadIdx.x][threadIdx.y + i * 8]);
    }
}

__global__ void k_bnprep(const float* __restrict__ g, const float* __restrict__ b,
                         const float* __restrict__ m, const float* __restrict__ v,
                         float* __restrict__ scale, float* __restrict__ shift) {
    int i = blockIdx.x * 256 + threadIdx.x;
    if (i >= VOCAB) return;
    float s = g[i] * rsqrtf(v[i] + 1e-3f);
    scale[i] = s;
    shift[i] = b[i] - m[i] * s;
}

// x_bf[s][b][e]  (time-major for the recurrence)
__global__ void k_embed(const int* __restrict__ ids, const float* __restrict__ emb,
                        uint16_t* __restrict__ x) {
    int id = blockIdx.x * 256 + threadIdx.x;   // 256*64*256 threads
    int e = id & 255, r = id >> 8;
    int b = r & 63,   s = r >> 6;
    int tok = ids[b * SS + s];
    x[id] = f2bf(emb[(size_t)tok * EE + e]);
}

__global__ void k_zero16(uint16_t* __restrict__ p, int n) {
    int i = blockIdx.x * 256 + threadIdx.x;
    if (i < n) p[i] = 0;
}

// ======================================================================
// GEMM inner loop over one A-segment: 4 M-tiles x 1 N-tile per wave.
// Software-pipelined with loop-carried pointers (+64B/step) so loads use
// immediate offsets and next-k fragments fly while current WMMAs execute.
// ======================================================================
template <int KI>
__device__ __forceinline__ void gemm_seg(const uint16_t* __restrict__ seg, int ld,
                                         const uint16_t* __restrict__ browIn,
                                         int l15, int hi, v8f acc[4]) {
    const uint16_t* p0 = seg + (size_t)l15 * ld + hi * 8;
    const uint16_t* p1 = p0 + (size_t)16 * ld;
    const uint16_t* p2 = p0 + (size_t)32 * ld;
    const uint16_t* p3 = p0 + (size_t)48 * ld;
    const uint16_t* pb = browIn + hi * 8;

    // prologue: fragment set for kk = 0
    v16bf Bf = pack_frag(*(const uint4*)pb, *(const uint4*)(pb + 16));
    v16bf A0 = pack_frag(*(const uint4*)p0, *(const uint4*)(p0 + 16));
    v16bf A1 = pack_frag(*(const uint4*)p1, *(const uint4*)(p1 + 16));
    v16bf A2 = pack_frag(*(const uint4*)p2, *(const uint4*)(p2 + 16));
    v16bf A3 = pack_frag(*(const uint4*)p3, *(const uint4*)(p3 + 16));

#pragma unroll 4
    for (int kk = 1; kk < KI; ++kk) {
        p0 += 32; p1 += 32; p2 += 32; p3 += 32; pb += 32;
        v16bf Bn = pack_frag(*(const uint4*)pb, *(const uint4*)(pb + 16));
        v16bf N0 = pack_frag(*(const uint4*)p0, *(const uint4*)(p0 + 16));
        v16bf N1 = pack_frag(*(const uint4*)p1, *(const uint4*)(p1 + 16));
        v16bf N2 = pack_frag(*(const uint4*)p2, *(const uint4*)(p2 + 16));
        v16bf N3 = pack_frag(*(const uint4*)p3, *(const uint4*)(p3 + 16));
        acc[0] = WMMA_BF16(A0, Bf, acc[0]);
        acc[1] = WMMA_BF16(A1, Bf, acc[1]);
        acc[2] = WMMA_BF16(A2, Bf, acc[2]);
        acc[3] = WMMA_BF16(A3, Bf, acc[3]);
        Bf = Bn; A0 = N0; A1 = N1; A2 = N2; A3 = N3;
    }
    acc[0] = WMMA_BF16(A0, Bf, acc[0]);
    acc[1] = WMMA_BF16(A1, Bf, acc[1]);
    acc[2] = WMMA_BF16(A2, Bf, acc[2]);
    acc[3] = WMMA_BF16(A3, Bf, acc[3]);
}

// gates: ru = sigmoid([a0,a1] @ wT^T + bias);  r-half -> rh = r*h, u-half -> ubuf
// one wave per N-tile (tn 0..127), all 4 M-tiles.  grid = 16 blocks x 256.
template <int KI0, int KI1>
__global__ void k_gru_gates(const uint16_t* __restrict__ a0, int ld0,
                            const uint16_t* __restrict__ a1,          // == h (stride HH)
                            const uint16_t* __restrict__ wT,          // [2H][K]
                            const float* __restrict__ bias,           // [2H]
                            uint16_t* __restrict__ rh,                // [B][H]
                            float* __restrict__ ubuf) {               // [B][H]
    int lane = threadIdx.x & 31, wave = threadIdx.x >> 5;
    int tn = blockIdx.x * 8 + wave;            // 0..127
    int l15 = lane & 15, hi = lane >> 4;
    const int Ktot = (KI0 + KI1) * 32;
    const uint16_t* brow = wT + (size_t)(tn * 16 + l15) * Ktot;

    v8f acc[4];
#pragma unroll
    for (int t = 0; t < 4; ++t) acc[t] = (v8f){0.f,0.f,0.f,0.f,0.f,0.f,0.f,0.f};

    gemm_seg<KI0>(a0, ld0, brow, l15, hi, acc);
    gemm_seg<KI1>(a1, HH, brow + KI0 * 32, l15, hi, acc);

    int n = tn * 16 + l15;
    float bv = bias[n];
#pragma unroll
    for (int t = 0; t < 4; ++t) {
#pragma unroll
        for (int j = 0; j < 8; ++j) {
            int m = t * 16 + j + hi * 8;
            float g = 1.f / (1.f + __expf(-(acc[t][j] + bv)));
            if (tn < 64) {                              // r half (n < H)
                float h = bf2f(a1[(size_t)m * HH + n]);
                rh[(size_t)m * HH + n] = f2bf(g * h);
            } else {                                    // u half
                ubuf[(size_t)m * HH + (n - HH)] = g;
            }
        }
    }
}

// candidate: c = tanh([a0, rh] @ cT^T + cb);  h = u*h + (1-u)*c  (in place)
// one wave per N-tile (tn 0..63).  grid = 8 blocks x 256.
template <int KI0, int KI1>
__global__ void k_gru_cand(const uint16_t* __restrict__ a0, int ld0,
                           const uint16_t* __restrict__ rh,
                           const uint16_t* __restrict__ wT,           // [H][K]
                           const float* __restrict__ bias,            // [H]
                           const float* __restrict__ ubuf,
                           uint16_t* __restrict__ h,                  // [B][H] in/out
                           uint16_t* __restrict__ outp) {             // out + s*H, or null
    int lane = threadIdx.x & 31, wave = threadIdx.x >> 5;
    int tn = blockIdx.x * 8 + wave;            // 0..63
    int l15 = lane & 15, hi = lane >> 4;
    const int Ktot = (KI0 + KI1) * 32;
    const uint16_t* brow = wT + (size_t)(tn * 16 + l15) * Ktot;

    v8f acc[4];
#pragma unroll
    for (int t = 0; t < 4; ++t) acc[t] = (v8f){0.f,0.f,0.f,0.f,0.f,0.f,0.f,0.f};

    gemm_seg<KI0>(a0, ld0, brow, l15, hi, acc);
    gemm_seg<KI1>(rh, HH, brow + KI0 * 32, l15, hi, acc);

    int n = tn * 16 + l15;
    float bv = bias[n];
#pragma unroll
    for (int t = 0; t < 4; ++t) {
#pragma unroll
        for (int j = 0; j < 8; ++j) {
            int m = t * 16 + j + hi * 8;
            float tt = acc[t][j] + bv;
            tt = fminf(fmaxf(tt, -15.f), 15.f);
            float e = __expf(2.f * tt);
            float c = (e - 1.f) / (e + 1.f);            // tanh
            float u = ubuf[(size_t)m * HH + n];
            float hold = bf2f(h[(size_t)m * HH + n]);
            float hnew = u * hold + (1.f - u) * c;
            h[(size_t)m * HH + n] = f2bf(hnew);
            if (outp) outp[(size_t)m * (SS * HH) + n] = f2bf(hnew);  // row b*S+s
        }
    }
}

// ======================================================================
// Output projection: logits = out @ W + b, fused BN; LDS-staged A tile.
// grid = (40, 1024), block = 256 (8 waves).  Each wave: 2 N-tiles,
// software-pipelined with carried pointers.
// ======================================================================
__global__ void k_out_gemm(const uint16_t* __restrict__ A,   // [16384][1024] bf16
                           const uint16_t* __restrict__ WT,  // [10000][1024] bf16
                           const float* __restrict__ sb,
                           const float* __restrict__ scale,
                           const float* __restrict__ shift,
                           float* __restrict__ out) {
    __shared__ uint16_t As[16 * HH];                     // 32 KB
    int tm = blockIdx.y;
    {   // cooperative 32KB copy (rows are contiguous)
        const uint4* src = (const uint4*)(A + (size_t)tm * 16 * HH);
        uint4* dst = (uint4*)As;
#pragma unroll
        for (int i = 0; i < 8; ++i)
            dst[threadIdx.x + i * 256] = src[threadIdx.x + i * 256];
    }
    __syncthreads();

    int lane = threadIdx.x & 31, wave = threadIdx.x >> 5;
    int tn0 = (blockIdx.x * 8 + wave) * 2;
    int tn1 = tn0 + 1;
    if (tn0 >= VOCAB / 16) return;                       // 625 tiles in N
    bool has1 = (tn1 < VOCAB / 16);
    int l15 = lane & 15, hi = lane >> 4;

    const uint16_t* pa  = As + (size_t)l15 * HH + hi * 8;
    const uint16_t* pb0 = WT + (size_t)(tn0 * 16 + l15) * HH + hi * 8;
    const uint16_t* pb1 = WT + (size_t)((has1 ? tn1 : tn0) * 16 + l15) * HH + hi * 8;

    v8f acc0 = {0.f,0.f,0.f,0.f,0.f,0.f,0.f,0.f};
    v8f acc1 = {0.f,0.f,0.f,0.f,0.f,0.f,0.f,0.f};

    v16bf Af = pack_frag(*(const uint4*)pa,  *(const uint4*)(pa + 16));
    v16bf B0 = pack_frag(*(const uint4*)pb0, *(const uint4*)(pb0 + 16));
    v16bf B1 = pack_frag(*(const uint4*)pb1, *(const uint4*)(pb1 + 16));

#pragma unroll 4
    for (int kk = 1; kk < HH / 32; ++kk) {
        pa += 32; pb0 += 32; pb1 += 32;
        v16bf An = pack_frag(*(const uint4*)pa,  *(const uint4*)(pa + 16));
        v16bf C0 = pack_frag(*(const uint4*)pb0, *(const uint4*)(pb0 + 16));
        v16bf C1 = pack_frag(*(const uint4*)pb1, *(const uint4*)(pb1 + 16));
        acc0 = WMMA_BF16(Af, B0, acc0);
        acc1 = WMMA_BF16(Af, B1, acc1);
        Af = An; B0 = C0; B1 = C1;
    }
    acc0 = WMMA_BF16(Af, B0, acc0);
    acc1 = WMMA_BF16(Af, B1, acc1);

    {
        int n = tn0 * 16 + l15;
        float sc = scale[n], sh = shift[n], bv = sb[n];
#pragma unroll
        for (int j = 0; j < 8; ++j) {
            int row = tm * 16 + j + hi * 8;
            out[(size_t)row * VOCAB + n] = (acc0[j] + bv) * sc + sh;
        }
    }
    if (has1) {
        int n = tn1 * 16 + l15;
        float sc = scale[n], sh = shift[n], bv = sb[n];
#pragma unroll
        for (int j = 0; j < 8; ++j) {
            int row = tm * 16 + j + hi * 8;
            out[(size_t)row * VOCAB + n] = (acc1[j] + bv) * sc + sh;
        }
    }
}

// ======================================================================
// Row softmax in place (one block per row of 10000)
// ======================================================================
__global__ void k_softmax(float* __restrict__ out) {
    __shared__ float red[256];
    float* p = out + (size_t)blockIdx.x * VOCAB;
    int t = threadIdx.x;

    float mx = -3.4e38f;
    for (int i = t; i < VOCAB; i += 256) mx = fmaxf(mx, p[i]);
    red[t] = mx; __syncthreads();
    for (int s = 128; s > 0; s >>= 1) {
        if (t < s) red[t] = fmaxf(red[t], red[t + s]);
        __syncthreads();
    }
    mx = red[0]; __syncthreads();

    float sum = 0.f;
    for (int i = t; i < VOCAB; i += 256) {
        float e = __expf(p[i] - mx);
        p[i] = e; sum += e;
    }
    red[t] = sum; __syncthreads();
    for (int s = 128; s > 0; s >>= 1) {
        if (t < s) red[t] += red[t + s];
        __syncthreads();
    }
    float inv = 1.f / red[0];
    for (int i = t; i < VOCAB; i += 256) p[i] *= inv;
}

// ======================================================================
// Host launcher
// ======================================================================
extern "C" void kernel_launch(void* const* d_in, const int* in_sizes, int n_in,
                              void* d_out, int out_size, void* d_ws, size_t ws_size,
                              hipStream_t stream) {
    (void)in_sizes; (void)n_in; (void)out_size; (void)ws_size;

    const int*   ids   = (const int*)  d_in[0];
    const float* emb   = (const float*)d_in[1];
    const float* gk0   = (const float*)d_in[2];
    const float* gb0   = (const float*)d_in[3];
    const float* ck0   = (const float*)d_in[4];
    const float* cb0   = (const float*)d_in[5];
    const float* gk1   = (const float*)d_in[6];
    const float* gb1   = (const float*)d_in[7];
    const float* ck1   = (const float*)d_in[8];
    const float* cb1   = (const float*)d_in[9];
    const float* smw   = (const float*)d_in[10];
    const float* smb   = (const float*)d_in[11];
    const float* bng   = (const float*)d_in[12];
    const float* bnb   = (const float*)d_in[13];
    const float* bnm   = (const float*)d_in[14];
    const float* bnv   = (const float*)d_in[15];

    char* ws = (char*)d_ws;
    uint16_t* gk0T = (uint16_t*)(ws + 0);               // 2048 x 1280
    uint16_t* ck0T = (uint16_t*)(ws + 5242880);         // 1024 x 1280
    uint16_t* gk1T = (uint16_t*)(ws + 7864320);         // 2048 x 2048
    uint16_t* ck1T = (uint16_t*)(ws + 16252928);        // 1024 x 2048
    uint16_t* WT   = (uint16_t*)(ws + 20447232);        // 10000 x 1024
    uint16_t* xbf  = (uint16_t*)(ws + 40927232);        // [S][B][E]
    uint16_t* h0   = (uint16_t*)(ws + 49315840);        // [B][H]
    uint16_t* h1   = (uint16_t*)(ws + 49446912);        // [B][H]
    uint16_t* rh   = (uint16_t*)(ws + 49577984);        // [B][H]
    float*    ubuf = (float*)   (ws + 49709056);        // [B][H]
    uint16_t* outb = (uint16_t*)(ws + 49971200);        // [B*S][H]
    float*    scl  = (float*)   (ws + 83525632);        // [V]
    float*    shf  = (float*)   (ws + 83565632);        // [V]

    float* probs = (float*)d_out;
    dim3 tb(32, 8);

    // ---- prep ----
    k_transpose_bf16<<<dim3(1280 / 32, 2048 / 32), tb, 0, stream>>>(gk0, gk0T, 1280, 2048);
    k_transpose_bf16<<<dim3(1280 / 32, 1024 / 32), tb, 0, stream>>>(ck0, ck0T, 1280, 1024);
    k_transpose_bf16<<<dim3(2048 / 32, 2048 / 32), tb, 0, stream>>>(gk1, gk1T, 2048, 2048);
    k_transpose_bf16<<<dim3(2048 / 32, 1024 / 32), tb, 0, stream>>>(ck1, ck1T, 2048, 1024);
    k_transpose_bf16<<<dim3(1024 / 32, (VOCAB + 31) / 32), tb, 0, stream>>>(smw, WT, 1024, VOCAB);
    k_bnprep<<<(VOCAB + 255) / 256, 256, 0, stream>>>(bng, bnb, bnm, bnv, scl, shf);
    k_embed<<<(SS * BB * EE) / 256, 256, 0, stream>>>(ids, emb, xbf);
    k_zero16<<<(2 * BB * HH + 255) / 256, 256, 0, stream>>>(h0, 2 * BB * HH); // h0+h1 contiguous

    // ---- recurrence ----
    for (int s = 0; s < SS; ++s) {
        const uint16_t* xs = xbf + (size_t)s * BB * EE;
        // layer 0:  K = 256 (x) + 1024 (h0)
        k_gru_gates<8, 32><<<16, 256, 0, stream>>>(xs, EE, h0, gk0T, gb0, rh, ubuf);
        k_gru_cand <8, 32><<< 8, 256, 0, stream>>>(xs, EE, rh, ck0T, cb0, ubuf, h0,
                                                   (uint16_t*)nullptr);
        // layer 1:  K = 1024 (h0) + 1024 (h1)
        k_gru_gates<32, 32><<<16, 256, 0, stream>>>(h0, HH, h1, gk1T, gb1, rh, ubuf);
        k_gru_cand <32, 32><<< 8, 256, 0, stream>>>(h0, HH, rh, ck1T, cb1, ubuf, h1,
                                                    outb + (size_t)s * HH);
    }

    // ---- projection + BN (fused) and softmax ----
    k_out_gemm<<<dim3(40, 1024), 256, 0, stream>>>(outb, WT, smb, scl, shf, probs);
    k_softmax<<<BB * SS, 256, 0, stream>>>(probs);
}
